// SnakeHead_74113955660005
// MI455X (gfx1250) — compile-verified, matrix operand order checked by reading, not run on hardware
//
#include <hip/hip_runtime.h>
#include <stdint.h>

typedef __bf16 bf16;
typedef __attribute__((ext_vector_type(16))) __bf16 v16bf;
typedef __attribute__((ext_vector_type(8)))  float  v8f;

#define B_   16
#define N_   1024
#define H_   256
#define W_   256
#define CF_  126
#define D_   128      // Cf + 2
#define CH_  512
#define ROWS_ (B_ * N_)   // 16384

// ---------------------------------------------------------------------------
// Weight packing: f32 [.., Cin, Cout] -> bf16 fragment-native layout
//   [tap][kchunk(CIN/32)][colTile(Cout/16)][lane(32)][16]
// so a wave's B fragment for one 16-col tile is a contiguous, coalesced
// 1KB chunk; each lane reads its 8 dwords as two b128 loads.
// Per ISA 7.12.2: lane(h=lane>>4,l16=lane&15) vgpr v holds K pair
//   kbase = 2v + 8h + (v>=4 ? 8 : 0), column = colTile*16 + l16.
// ---------------------------------------------------------------------------
template <int CIN, int TAPS, int LAYERS>
__global__ __launch_bounds__(256)
void pack_frag_kernel(const float* __restrict__ src, bf16* __restrict__ pk) {
    const int total = LAYERS * TAPS * (CIN / 32) * (CH_ / 16) * 32 * 16;
    int idx = blockIdx.x * 256 + threadIdx.x;
    if (idx >= total) return;
    const int p    = idx & 1;
    const int v    = (idx >> 1) & 7;
    const int lane = (idx >> 4) & 31;
    const int ct   = (idx >> 9) & (CH_ / 16 - 1);   // 0..31
    int rest = idx >> 14;
    const int kcI = rest % (CIN / 32);  rest /= (CIN / 32);
    const int tap = rest % TAPS;
    const int l   = rest / TAPS;

    const int h     = lane >> 4;
    const int l16   = lane & 15;
    const int kbase = 2 * v + 8 * h + ((v >= 4) ? 8 : 0);
    const int k     = kcI * 32 + kbase + p;
    const int col   = ct * 16 + l16;
    // src layout: [layer][tap][CIN][CH_]
    pk[idx] = (bf16)src[((size_t)((l * TAPS + tap) * CIN) + k) * CH_ + col];
}

// ---------------------------------------------------------------------------
// Bilinear gather + vertex concat -> x0 bf16 [16384][128]
// ---------------------------------------------------------------------------
__global__ __launch_bounds__(128)
void gather_kernel(const float* __restrict__ verts,
                   const float* __restrict__ fm,
                   bf16* __restrict__ x0) {
    const int pid = blockIdx.x;          // 0..16383
    const int b   = pid >> 10;
    const int c   = threadIdx.x;         // 0..127
    const float vy = verts[(size_t)pid * 2 + 0];
    const float vx = verts[(size_t)pid * 2 + 1];
    if (c >= CF_) {
        x0[(size_t)pid * D_ + c] = (bf16)((c == CF_) ? vy : vx);
        return;
    }
    const float cy = (vy + 1.0f) * ((H_ - 1) * 0.5f);
    const float cx = (vx + 1.0f) * ((W_ - 1) * 0.5f);
    const float fy = floorf(cy), fx = floorf(cx);
    const float wy = cy - fy,    wx = cx - fx;
    const int y0 = (int)fy, xx0 = (int)fx;
    float acc = 0.0f;
#pragma unroll
    for (int dy = 0; dy < 2; ++dy)
#pragma unroll
        for (int dx = 0; dx < 2; ++dx) {
            const int yy = y0 + dy, xx = xx0 + dx;
            if (yy >= 0 && yy < H_ && xx >= 0 && xx < W_) {
                const float w = (dy ? wy : 1.0f - wy) * (dx ? wx : 1.0f - wx);
                acc += w * fm[(((size_t)b * H_ + yy) * W_ + xx) * CF_ + c];
            }
        }
    x0[(size_t)pid * D_ + c] = (bf16)acc;
}

// ---------------------------------------------------------------------------
// Dilated conv as WMMA GEMM, LDS-free (everything L2-resident).
// Block: 256 thr = 8 waves as 4(M) x 2(N); block tile 128M x 128N.
// Wave tile: 32M x 64N = 2 A-frags x 4 B-tiles = 8 accumulators.
// A frags: two b128 loads per lane straight from row-major activations
// (the ISA K-pair layout makes each lane's half-fragment contiguous).
// B frags: two b128 loads per lane from fragment-native packed weights.
// ---------------------------------------------------------------------------
template <int CIN, int NTAPS>
__global__ __launch_bounds__(256)
void conv_wmma_kernel(const bf16* __restrict__ X,
                      const bf16* __restrict__ Wp,
                      const float* __restrict__ bias,
                      bf16* __restrict__ Y,
                      int Cout, int dil, int do_relu) {
    const int tid  = threadIdx.x;
    const int wave = tid >> 5;
    const int lane = tid & 31;
    const int h    = lane >> 4;
    const int l16  = lane & 15;
    const int wm   = wave >> 1;          // 0..3 along M
    const int wn   = wave & 1;           // 0..1 along N
    const int rowBase = blockIdx.y * 128 + wm * 32;
    const int colBase = blockIdx.x * 128 + wn * 64;
    const int ct0     = colBase >> 4;    // first 16-col tile index
    const int nTiles  = Cout >> 4;

    v8f acc[2][4];
#pragma unroll
    for (int f = 0; f < 2; ++f)
#pragma unroll
        for (int j = 0; j < 4; ++j)
#pragma unroll
            for (int i = 0; i < 8; ++i) acc[f][j][i] = 0.0f;

    union Frag { v16bf v; uint4 q[2]; };

    const int r0  = rowBase + l16;       // rows for frag0 / frag1
    const int r1  = r0 + 16;
    const int nl0 = r0 & (N_ - 1);
    const int nl1 = r1 & (N_ - 1);

#pragma unroll
    for (int t = 0; t < NTAPS; ++t) {
        const int noff = (t - (NTAPS >> 1)) * dil;
        const bool v0 = (unsigned)(nl0 + noff) < (unsigned)N_;
        const bool v1 = (unsigned)(nl1 + noff) < (unsigned)N_;
        const bf16* wtap = Wp + (size_t)t * (CIN / 32) * nTiles * 512
                              + (size_t)ct0 * 512 + lane * 16;
        for (int kcI = 0; kcI < CIN / 32; ++kcI) {
            const int kc = kcI * 32;
            Frag a0, a1;
            a0.q[0] = (uint4){0u,0u,0u,0u}; a0.q[1] = (uint4){0u,0u,0u,0u};
            a1.q[0] = (uint4){0u,0u,0u,0u}; a1.q[1] = (uint4){0u,0u,0u,0u};
            if (v0) {
                const bf16* p0 = X + (size_t)(r0 + noff) * CIN + kc + 8 * h;
                a0.q[0] = *reinterpret_cast<const uint4*>(p0);
                a0.q[1] = *reinterpret_cast<const uint4*>(p0 + 16);
            }
            if (v1) {
                const bf16* p1 = X + (size_t)(r1 + noff) * CIN + kc + 8 * h;
                a1.q[0] = *reinterpret_cast<const uint4*>(p1);
                a1.q[1] = *reinterpret_cast<const uint4*>(p1 + 16);
            }
            const bf16* wk = wtap + (size_t)kcI * nTiles * 512;
            if (kcI + 1 < CIN / 32)
                __builtin_prefetch(wk + (size_t)nTiles * 512, 0, 0);
#pragma unroll
            for (int j = 0; j < 4; ++j) {
                Frag b;
                const uint4* bp = reinterpret_cast<const uint4*>(wk + (size_t)j * 512);
                b.q[0] = bp[0];
                b.q[1] = bp[1];
                acc[0][j] = __builtin_amdgcn_wmma_f32_16x16x32_bf16(
                    false, a0.v, false, b.v, (short)0, acc[0][j], false, false);
                acc[1][j] = __builtin_amdgcn_wmma_f32_16x16x32_bf16(
                    false, a1.v, false, b.v, (short)0, acc[1][j], false, false);
            }
        }
    }

    // epilogue: bias + relu + bf16 store (C layout: VGPR i -> M = i + 8h)
#pragma unroll
    for (int f = 0; f < 2; ++f)
#pragma unroll
        for (int j = 0; j < 4; ++j) {
            const int col = colBase + j * 16 + l16;
            const float bv = bias[col];
#pragma unroll
            for (int i = 0; i < 8; ++i) {
                float val = acc[f][j][i] + bv;
                if (do_relu) val = fmaxf(val, 0.0f);
                const int m = rowBase + f * 16 + i + 8 * h;
                Y[(size_t)m * Cout + col] = (bf16)val;
            }
        }
}

// ---------------------------------------------------------------------------
// Head: [16384,512] bf16 x [512,2] f32 -> f32 [16384,2]; wave per point.
// ---------------------------------------------------------------------------
__global__ __launch_bounds__(256)
void head_kernel(const bf16* __restrict__ X, const float* __restrict__ woff,
                 float* __restrict__ out) {
    const int wave = threadIdx.x >> 5;
    const int lane = threadIdx.x & 31;
    const int p = blockIdx.x * 8 + wave;
    float a0 = 0.0f, a1 = 0.0f;
    const bf16* xr = X + (size_t)p * CH_ + lane * 16;
#pragma unroll
    for (int k = 0; k < 16; ++k) {
        const int c = lane * 16 + k;
        const float xv = (float)xr[k];
        a0 += xv * woff[c * 2 + 0];
        a1 += xv * woff[c * 2 + 1];
    }
    for (int off = 16; off > 0; off >>= 1) {
        a0 += __shfl_down(a0, off);
        a1 += __shfl_down(a1, off);
    }
    if (lane == 0) {
        out[(size_t)p * 2 + 0] = a0;
        out[(size_t)p * 2 + 1] = a1;
    }
}

// ---------------------------------------------------------------------------
extern "C" void kernel_launch(void* const* d_in, const int* in_sizes, int n_in,
                              void* d_out, int out_size, void* d_ws, size_t ws_size,
                              hipStream_t stream) {
    (void)in_sizes; (void)n_in; (void)out_size; (void)ws_size;
    const float* verts = (const float*)d_in[0];
    const float* fm    = (const float*)d_in[1];
    const float* w0    = (const float*)d_in[2];
    const float* b0    = (const float*)d_in[3];
    const float* ws    = (const float*)d_in[4];
    const float* bs    = (const float*)d_in[5];
    const float* woff  = (const float*)d_in[6];
    float* out = (float*)d_out;

    char* base = (char*)d_ws;
    size_t off = 0;
    auto salloc = [&](size_t bytes) -> void* {
        void* p = base + off;
        off += (bytes + 255) & ~(size_t)255;
        return p;
    };
    bf16* pk0 = (bf16*)salloc((size_t)D_ * CH_ * sizeof(bf16));
    bf16* pkw = (bf16*)salloc((size_t)6 * 3 * CH_ * CH_ * sizeof(bf16));
    bf16* x0  = (bf16*)salloc((size_t)ROWS_ * D_ * sizeof(bf16));
    bf16* xA  = (bf16*)salloc((size_t)ROWS_ * CH_ * sizeof(bf16));
    bf16* xB  = (bf16*)salloc((size_t)ROWS_ * CH_ * sizeof(bf16));

    // 1) pack weights to bf16, fragment-native layout
    {
        const int total0 = D_ * CH_;
        pack_frag_kernel<D_, 1, 1><<<(total0 + 255) / 256, 256, 0, stream>>>(w0, pk0);
        const int totalw = 6 * 3 * CH_ * CH_;
        pack_frag_kernel<CH_, 3, 6><<<(totalw + 255) / 256, 256, 0, stream>>>(ws, pkw);
    }
    // 2) bilinear gather + concat
    gather_kernel<<<ROWS_, 128, 0, stream>>>(verts, fm, x0);

    // 3) input 1x1 conv (128 -> 512), relu
    {
        dim3 grid(CH_ / 128, ROWS_ / 128);
        conv_wmma_kernel<D_, 1><<<grid, 256, 0, stream>>>(
            x0, pk0, b0, xA, CH_, /*dil=*/1, /*relu=*/1);
    }
    // 4) six dilated K=3 convs, relu, ping-pong
    {
        static const int DIL[6] = {1, 3, 9, 9, 3, 1};
        dim3 grid(CH_ / 128, ROWS_ / 128);
        for (int l = 0; l < 6; ++l) {
            const bf16* in = (l & 1) ? xB : xA;
            bf16*       ot = (l & 1) ? xA : xB;
            const bf16* wp = pkw + (size_t)l * 3 * CH_ * CH_;
            conv_wmma_kernel<CH_, 3><<<grid, 256, 0, stream>>>(
                in, wp, bs + l * CH_, ot, CH_, DIL[l], /*relu=*/1);
        }
    }
    // 5) final 512 -> 2 head (layer 5 wrote xA)
    head_kernel<<<ROWS_ / 8, 256, 0, stream>>>(xA, woff, out);
}
